// OLMoE1B7B0924Synapse_12275016532078
// MI455X (gfx1250) — matrix operand order repaired and checked by reading, not compiled, not run on hardware
//
#include <hip/hip_runtime.h>
#include <hip/hip_bf16.h>
#include <math.h>

// Problem constants (OLMoE block): D=2048, H=16, HD=128, E=64, K=8, DFF=1024, T=256
#define DM 2048
#define NH 16
#define HD 128
#define NE 64
#define TK 8
#define DFF 1024
#define TT 256
#define EPSF 1e-5f
#define ATT_SCALE 0.08838834764831845f

typedef __attribute__((ext_vector_type(16))) __bf16 v16bf;
typedef __attribute__((ext_vector_type(8)))  float  v8f;
typedef __attribute__((ext_vector_type(4)))  int    v4i;
typedef __attribute__((ext_vector_type(4)))  unsigned u32x4;
typedef __attribute__((ext_vector_type(8)))  int    i32x8;
typedef __attribute__((ext_vector_type(4)))  int    i32x4;

union BF16x16 { v16bf v; unsigned u[8]; };
union BFPair  { __bf16 h[2]; unsigned u; };

#if defined(__has_builtin)
#if __has_builtin(__builtin_amdgcn_global_load_async_to_lds_b128)
#define USE_ASYNC_LDS 1
#endif
#if __has_builtin(__builtin_amdgcn_tensor_load_to_lds)
#define USE_TDM 1
#endif
#endif

// 16B global -> LDS copy. CDNA5 path: async op tracked by ASYNCcnt.
__device__ __forceinline__ void cp16(void* lds, const float* g) {
#ifdef USE_ASYNC_LDS
  __builtin_amdgcn_global_load_async_to_lds_b128(
      (__attribute__((address_space(1))) v4i*)const_cast<float*>(g),
      (__attribute__((address_space(3))) v4i*)lds, 0, 0);
#else
  *(float4*)lds = *(const float4*)g;
#endif
}

__device__ __forceinline__ void wait_async_lds() {
#ifdef USE_ASYNC_LDS
#if __has_builtin(__builtin_amdgcn_s_wait_asynccnt)
  __builtin_amdgcn_s_wait_asynccnt(0);
#else
  asm volatile("s_wait_asynccnt 0x0" ::: "memory");
#endif
#endif
}

__device__ __forceinline__ void wait_tensor() {
#ifdef USE_TDM
#if __has_builtin(__builtin_amdgcn_s_wait_tensorcnt)
  __builtin_amdgcn_s_wait_tensorcnt(0);
#else
  asm volatile("s_wait_tensorcnt 0x0" ::: "memory");
#endif
#endif
}

// TDM: DMA a (rows x 32 f32) tile, global row stride = stride_elts (f32 units),
// into LDS with 36-dword padded rows (pad_interval=32 dwords, pad_amount=4 dwords).
// Issue from ONE wave only; completion via TENSORcnt.
__device__ __forceinline__ void tdm_load_2d(const float* g, float* lds,
                                            unsigned rows, unsigned stride_elts) {
#ifdef USE_TDM
  const unsigned long long ga = (unsigned long long)(size_t)g;
  const unsigned ldsa =
      (unsigned)(size_t)(__attribute__((address_space(3))) char*)(void*)lds;
  u32x4 g0;
  g0[0] = 1u;                                             // count=1 (user D#)
  g0[1] = ldsa;                                           // lds_addr (bytes)
  g0[2] = (unsigned)ga;                                   // global_addr[31:0]
  g0[3] = (unsigned)((ga >> 32) & 0x01ffffffu) | 0x80000000u; // addr[56:32]|type=2
  i32x8 g1;
  g1[0] = (int)((2u << 16) | (1u << 20) | (4u << 22) | (3u << 25));
         // data_size=4B | pad_enable | pad_interval: 32 dwords | pad_amount: 4 dwords
  g1[1] = (int)(32u << 16);                               // tensor_dim0 = 32 (lo16)
  g1[2] = (int)((rows & 0xffffu) << 16);                  // tensor_dim1 (lo16)
  g1[3] = (int)((rows >> 16) | (32u << 16));              // dim1 hi | tile_dim0 = 32
  g1[4] = (int)(rows & 0xffffu);                          // tile_dim1 = rows; dim2=0
  g1[5] = (int)stride_elts;                               // tensor_dim0_stride lo32
  g1[6] = 0;
  g1[7] = 0;
  i32x4 gz = {0, 0, 0, 0};
  i32x8 gz8 = {0, 0, 0, 0, 0, 0, 0, 0};
  __builtin_amdgcn_tensor_load_to_lds(g0, g1, gz, gz, gz8, 0);
#else
  const int lane = threadIdx.x & 31;                      // wave-0 cooperative copy
  for (unsigned r = lane; r < rows; r += 32)
#pragma unroll
    for (int c = 0; c < 32; c += 4)
      *(float4*)&lds[r * 36 + c] = *(const float4*)&g[(size_t)r * stride_elts + c];
#endif
}

__device__ __forceinline__ bool is_wave0() {
  return __builtin_amdgcn_readfirstlane((int)(threadIdx.x >> 5)) == 0;
}

__device__ __forceinline__ v8f wmma_bf16(v16bf a, v16bf b, v8f c) {
  return __builtin_amdgcn_wmma_f32_16x16x32_bf16(false, a, false, b, (short)0, c,
                                                 false, false);
}

// pack two f32 -> packed bf16 dword
__device__ __forceinline__ unsigned cvt2(float a, float b) {
#if defined(__has_builtin)
#if __has_builtin(__builtin_amdgcn_cvt_pk_bf16_f32)
  typedef __attribute__((ext_vector_type(2))) __bf16 v2bf;
  union { v2bf v; unsigned u; } r;
  r.v = __builtin_amdgcn_cvt_pk_bf16_f32(a, b);
  return r.u;
#else
  BFPair p; p.h[0] = (__bf16)a; p.h[1] = (__bf16)b; return p.u;
#endif
#else
  BFPair p; p.h[0] = (__bf16)a; p.h[1] = (__bf16)b; return p.u;
#endif
}

// ---- WMMA fragment loaders (ISA 7.12.2 wave32 layouts) ----
__device__ __forceinline__ v16bf load_a_frag_lds(const __bf16* base, int stride, int lane) {
  const int m = lane & 15, half = lane >> 4;
  const __bf16* row = base + m * stride;
  BF16x16 f;
#pragma unroll
  for (int v = 0; v < 8; ++v) {
    int pi = (v < 4 ? v : v + 4) + half * 4;
    f.u[v] = *(const unsigned*)(row + 2 * pi);
  }
  return f.v;
}
__device__ __forceinline__ v16bf load_b_frag_lds(const __bf16* base, int stride, int lane) {
  const int n = lane & 15, half = lane >> 4;
  const __bf16* row = base + n * stride;
  BF16x16 f;
#pragma unroll
  for (int v = 0; v < 8; ++v) {
    int pi = (v < 4 ? v : v + 4) + half * 4;
    f.u[v] = *(const unsigned*)(row + 2 * pi);
  }
  return f.v;
}
__device__ __forceinline__ v16bf load_a_frag_f32(const float* base, int stride, int lane) {
  const int m = lane & 15, half = lane >> 4;
  const float* row = base + m * stride;
  BF16x16 f;
#pragma unroll
  for (int v = 0; v < 8; ++v) {
    int pi = (v < 4 ? v : v + 4) + half * 4;
    f.u[v] = cvt2(row[2 * pi], row[2 * pi + 1]);
  }
  return f.v;
}
__device__ __forceinline__ v16bf load_b_frag_f32(const float* base, int stride, int lane) {
  const int n = lane & 15, half = lane >> 4;
  const float* row = base + n * stride;
  BF16x16 f;
#pragma unroll
  for (int v = 0; v < 8; ++v) {
    int pi = (v < 4 ? v : v + 4) + half * 4;
    f.u[v] = cvt2(row[2 * pi], row[2 * pi + 1]);
  }
  return f.v;
}
__device__ __forceinline__ v16bf load_a_frag_gf32(const float* base, int stride, int lane,
                                                  float scale) {
  const int m = lane & 15, half = lane >> 4;
  const float* row = base + (size_t)m * stride;
  BF16x16 f;
#pragma unroll
  for (int v = 0; v < 8; ++v) {
    int pi = (v < 4 ? v : v + 4) + half * 4;
    f.u[v] = cvt2(row[2 * pi] * scale, row[2 * pi + 1] * scale);
  }
  return f.v;
}

// ---------------- RMSNorm ----------------
__global__ __launch_bounds__(256) void rmsnorm_kernel(const float* __restrict__ in,
                                                      const float* __restrict__ w,
                                                      float* __restrict__ out) {
  __shared__ float red[8];
  const int t = blockIdx.x, tid = threadIdx.x;
  const float* row = in + (size_t)t * DM;
  float ss = 0.f;
#pragma unroll
  for (int j = 0; j < 8; ++j) { float x = row[tid + 256 * j]; ss += x * x; }
#pragma unroll
  for (int o = 16; o > 0; o >>= 1) ss += __shfl_xor(ss, o, 32);
  if ((tid & 31) == 0) red[tid >> 5] = ss;
  __syncthreads();
  float tot = 0.f;
#pragma unroll
  for (int i = 0; i < 8; ++i) tot += red[i];
  const float scale = rsqrtf(tot / (float)DM + EPSF);
#pragma unroll
  for (int j = 0; j < 8; ++j) {
    int i = tid + 256 * j;
    out[(size_t)t * DM + i] = row[i] * scale * w[i];
  }
}

// ------- NT GEMM: A staged via per-lane async, W via TDM; double-buffered -------
__global__ __launch_bounds__(256) void gemm_nt_kernel(const float* __restrict__ A,
                                                      const float* __restrict__ W,
                                                      const float* __restrict__ R,
                                                      float* __restrict__ C,
                                                      int M, int N, int K) {
  __shared__ __align__(16) float Af[2][64][36];
  __shared__ __align__(16) float Wf[2][128][36];
  const int tid = threadIdx.x, lane = tid & 31, wv = tid >> 5;
  const bool w0 = is_wave0();
  const int mblk = blockIdx.y * 64, nblk = blockIdx.x * 128;
  const int wm = (wv >> 2) * 32, wn = (wv & 3) * 32;
  v8f acc[2][2];
#pragma unroll
  for (int i = 0; i < 2; ++i)
#pragma unroll
    for (int j = 0; j < 2; ++j) acc[i][j] = {};

  auto issueA = [&](int buf, int k0) {
#pragma unroll
    for (int j = 0; j < 2; ++j) {
      int ch = tid + 256 * j, r = ch >> 3, c4 = (ch & 7) * 4;
      cp16(&Af[buf][r][c4], &A[(size_t)(mblk + r) * K + k0 + c4]);
    }
  };

  issueA(0, 0);
  if (w0) tdm_load_2d(&W[(size_t)nblk * K], &Wf[0][0][0], 128, (unsigned)K);
  const int KT = K >> 5;
  for (int kt = 0; kt < KT; ++kt) {
    const int cur = kt & 1;
    wait_async_lds();
    if (w0) wait_tensor();
    __syncthreads();
    if (kt + 1 < KT) {
      const int k0n = (kt + 1) << 5;
      issueA(cur ^ 1, k0n);
      if (w0)
        tdm_load_2d(&W[(size_t)nblk * K + k0n], &Wf[cur ^ 1][0][0], 128, (unsigned)K);
    }
    v16bf a0 = load_a_frag_f32(&Af[cur][wm][0], 36, lane);
    v16bf a1 = load_a_frag_f32(&Af[cur][wm + 16][0], 36, lane);
    v16bf b0 = load_b_frag_f32(&Wf[cur][wn][0], 36, lane);
    v16bf b1 = load_b_frag_f32(&Wf[cur][wn + 16][0], 36, lane);
    acc[0][0] = wmma_bf16(a0, b0, acc[0][0]);
    acc[0][1] = wmma_bf16(a0, b1, acc[0][1]);
    acc[1][0] = wmma_bf16(a1, b0, acc[1][0]);
    acc[1][1] = wmma_bf16(a1, b1, acc[1][1]);
  }
  const int half = lane >> 4, nl = lane & 15;
#pragma unroll
  for (int i = 0; i < 2; ++i)
#pragma unroll
    for (int j = 0; j < 2; ++j)
#pragma unroll
      for (int r = 0; r < 8; ++r) {
        int row = mblk + wm + i * 16 + r + 8 * half;
        int col = nblk + wn + j * 16 + nl;
        float v = acc[i][j][r];
        if (R) v += R[(size_t)row * N + col];
        C[(size_t)row * N + col] = v;
      }
}

// ---------------- RoPE ----------------
__global__ __launch_bounds__(1024) void rope_kernel(float* __restrict__ q,
                                                    float* __restrict__ k) {
  const int t = blockIdx.x;
  const int h = threadIdx.x >> 6, i = threadIdx.x & 63;
  const float inv = __powf(10000.0f, -(float)i / 64.0f);
  const float a = (float)t * inv;
  float sa, ca;
  __sincosf(a, &sa, &ca);
  const size_t i1 = (size_t)t * DM + h * HD + i, i2 = i1 + 64;
  float q1 = q[i1], q2 = q[i2];
  q[i1] = q1 * ca - q2 * sa;
  q[i2] = q1 * sa + q2 * ca;
  float k1 = k[i1], k2 = k[i2];
  k[i1] = k1 * ca - k2 * sa;
  k[i2] = k1 * sa + k2 * ca;
}

// ---------------- Flash attention ----------------
__global__ __launch_bounds__(256) void attn_kernel(const float* __restrict__ q,
                                                   const float* __restrict__ k,
                                                   const float* __restrict__ v,
                                                   float* __restrict__ ctx) {
  __shared__ __align__(16) __bf16 Ks[32][130];
  __shared__ __align__(16) __bf16 Vs[128][34];
  __shared__ __align__(16) __bf16 Ps[8][16][34];
  const int h = blockIdx.x & 15, qh = blockIdx.x >> 4;
  const int tid = threadIdx.x, lane = tid & 31, wv = tid >> 5;
  const int qrow0 = (qh * 8 + wv) * 16;
  const int half = lane >> 4, nl = lane & 15;

  v16bf qf[4];
#pragma unroll
  for (int c = 0; c < 4; ++c)
    qf[c] = load_a_frag_gf32(q + (size_t)qrow0 * DM + h * HD + c * 32, DM, lane, ATT_SCALE);

  v8f acc[8];
#pragma unroll
  for (int ht = 0; ht < 8; ++ht) acc[ht] = {};
  float rowmax[8], rowsum[8];
#pragma unroll
  for (int r = 0; r < 8; ++r) { rowmax[r] = -1e30f; rowsum[r] = 0.f; }

  const int jmax = (qh == 0) ? 4 : 8;
  for (int j = 0; j < jmax; ++j) {
#pragma unroll
    for (int jj = 0; jj < 16; ++jj) {
      int e = tid + 256 * jj;
      int key = e >> 7, hd = e & 127;
      Ks[key][hd] = (__bf16)k[(size_t)(j * 32 + key) * DM + h * HD + hd];
      Vs[hd][key] = (__bf16)v[(size_t)(j * 32 + key) * DM + h * HD + hd];
    }
    __syncthreads();

    v8f s[2];
#pragma unroll
    for (int st = 0; st < 2; ++st) {
      s[st] = {};
#pragma unroll
      for (int c = 0; c < 4; ++c)
        s[st] = wmma_bf16(qf[c], load_b_frag_lds(&Ks[st * 16][c * 32], 130, lane), s[st]);
    }
#pragma unroll
    for (int st = 0; st < 2; ++st)
#pragma unroll
      for (int r = 0; r < 8; ++r) {
        int mrow = qrow0 + r + 8 * half;
        int ncol = j * 32 + st * 16 + nl;
        if (ncol > mrow) s[st][r] = -3.0e38f;
      }
#pragma unroll
    for (int r = 0; r < 8; ++r) {
      float bmax = fmaxf(s[0][r], s[1][r]);
#pragma unroll
      for (int o = 1; o < 16; o <<= 1) bmax = fmaxf(bmax, __shfl_xor(bmax, o, 32));
      float mnew = fmaxf(rowmax[r], bmax);
      float scl = __expf(rowmax[r] - mnew);
      rowmax[r] = mnew;
      float p0 = __expf(s[0][r] - mnew);
      float p1 = __expf(s[1][r] - mnew);
      float ps = p0 + p1;
#pragma unroll
      for (int o = 1; o < 16; o <<= 1) ps += __shfl_xor(ps, o, 32);
      rowsum[r] = rowsum[r] * scl + ps;
#pragma unroll
      for (int ht = 0; ht < 8; ++ht) acc[ht][r] *= scl;
      Ps[wv][r + 8 * half][nl] = (__bf16)p0;
      Ps[wv][r + 8 * half][16 + nl] = (__bf16)p1;
    }
    asm volatile("s_wait_dscnt 0" ::: "memory");
    v16bf pf = load_a_frag_lds(&Ps[wv][0][0], 34, lane);
#pragma unroll
    for (int ht = 0; ht < 8; ++ht)
      acc[ht] = wmma_bf16(pf, load_b_frag_lds(&Vs[ht * 16][0], 34, lane), acc[ht]);
    __syncthreads();
  }
#pragma unroll
  for (int ht = 0; ht < 8; ++ht)
#pragma unroll
    for (int r = 0; r < 8; ++r) {
      int row = qrow0 + r + 8 * half;
      int col = h * HD + ht * 16 + nl;
      ctx[(size_t)row * DM + col] = acc[ht][r] / rowsum[r];
    }
}

// ---------------- Router ----------------
__global__ __launch_bounds__(64) void router_kernel(const float* __restrict__ x3,
                                                    const float* __restrict__ gw,
                                                    int* __restrict__ sel_e,
                                                    float* __restrict__ sel_w) {
  __shared__ float xrow[DM];
  __shared__ float pr[NE];
  const int t = blockIdx.x, e = threadIdx.x;
#pragma unroll
  for (int j = 0; j < 32; ++j) xrow[e + 64 * j] = x3[(size_t)t * DM + e + 64 * j];
  __syncthreads();
  float acc = 0.f;
  for (int kk = 0; kk < DM; ++kk) acc += xrow[kk] * gw[(size_t)e * DM + kk];
  pr[e] = acc;
  __syncthreads();
  float m = -1e30f;
  for (int i = 0; i < NE; ++i) m = fmaxf(m, pr[i]);
  float p = __expf(acc - m);
  __syncthreads();
  pr[e] = p;
  __syncthreads();
  float s = 0.f;
  for (int i = 0; i < NE; ++i) s += pr[i];
  __syncthreads();
  pr[e] = p / s;
  __syncthreads();
  if (e == 0) {
    for (int kk = 0; kk < TK; ++kk) {
      float best = -1.f; int bi = 0;
      for (int i = 0; i < NE; ++i)
        if (pr[i] > best) { best = pr[i]; bi = i; }
      sel_e[t * TK + kk] = bi;
      sel_w[t * TK + kk] = best;
      pr[bi] = -2.f;
    }
  }
}

// ---------------- Deterministic per-expert token lists ----------------
__global__ __launch_bounds__(64) void build_lists_kernel(const int* __restrict__ sel_e,
                                                         const float* __restrict__ sel_w,
                                                         int* __restrict__ cnt,
                                                         int* __restrict__ off,
                                                         int* __restrict__ ptok,
                                                         float* __restrict__ pw,
                                                         int* __restrict__ tpair) {
  const int e = threadIdx.x;
  int c = 0;
  for (int t = 0; t < TT; ++t)
    for (int s = 0; s < TK; ++s)
      if (sel_e[t * TK + s] == e) ++c;
  cnt[e] = c;
  __syncthreads();
  if (e == 0) {
    int run = 0;
    for (int i = 0; i < NE; ++i) { off[i] = run; run += cnt[i]; }
  }
  __syncthreads();
  int idx = off[e];
  for (int t = 0; t < TT; ++t)
    for (int s = 0; s < TK; ++s)
      if (sel_e[t * TK + s] == e) {
        ptok[idx] = t;
        pw[idx] = sel_w[t * TK + s];
        tpair[t * TK + s] = idx;
        ++idx;
      }
}

// -------- MoE gate_up: gathered A, W via two TDM descriptors, SiLU fused in regs ------
__global__ __launch_bounds__(256) void moe_up_kernel(const float* __restrict__ x3,
                                                     const float* __restrict__ gup,
                                                     const int* __restrict__ cnt,
                                                     const int* __restrict__ off,
                                                     const int* __restrict__ ptok,
                                                     const float* __restrict__ pw,
                                                     float* __restrict__ act) {
  const int e = blockIdx.x, f0 = blockIdx.y * 128;
  const int count = cnt[e];
  if (count == 0) return;
  const int base = off[e];
  __shared__ __align__(16) float Ak[64][36];
  __shared__ __align__(16) float Wk[256][36];        // rows 0-127 gate, 128-255 up
  const int tid = threadIdx.x, lane = tid & 31, wv = tid >> 5;
  const bool w0 = is_wave0();
  const int half = lane >> 4, nl = lane & 15;
  const size_t wbase = (size_t)e * 2 * DFF * DM;

  for (int mt = 0; mt * 64 < count; ++mt) {
    v8f accg[4], accu[4];
#pragma unroll
    for (int i = 0; i < 4; ++i) { accg[i] = {}; accu[i] = {}; }
    for (int k0 = 0; k0 < DM; k0 += 32) {
#pragma unroll
      for (int j = 0; j < 8; ++j) {                  // gathered A rows, zero-padded
        int el = tid + 256 * j, r = el >> 5, c = el & 31;
        int ri = mt * 64 + r;
        Ak[r][c] = (ri < count) ? x3[(size_t)ptok[base + ri] * DM + k0 + c] : 0.f;
      }
      if (w0) {                                      // TDM: gate rows + up rows
        tdm_load_2d(&gup[wbase + (size_t)f0 * DM + k0], &Wk[0][0], 128, DM);
        tdm_load_2d(&gup[wbase + (size_t)(DFF + f0) * DM + k0], &Wk[128][0], 128, DM);
        wait_tensor();
      }
      __syncthreads();
      v16bf bg = load_b_frag_f32(&Wk[wv * 16][0], 36, lane);
      v16bf bu = load_b_frag_f32(&Wk[128 + wv * 16][0], 36, lane);
#pragma unroll
      for (int i = 0; i < 4; ++i) {
        v16bf a = load_a_frag_f32(&Ak[i * 16][0], 36, lane);
        accg[i] = wmma_bf16(a, bg, accg[i]);
        accu[i] = wmma_bf16(a, bu, accu[i]);
      }
      __syncthreads();
    }
#pragma unroll
    for (int i = 0; i < 4; ++i)
#pragma unroll
      for (int r = 0; r < 8; ++r) {
        int m = i * 16 + r + 8 * half;
        int ri = mt * 64 + m;
        if (ri < count) {
          float g = accg[i][r], u = accu[i][r];
          float a = (g / (1.0f + __expf(-g))) * u * pw[base + ri];
          act[(size_t)(base + ri) * DFF + f0 + wv * 16 + nl] = a;
        }
      }
  }
}

// -------- MoE down: manual A (zero-padded), W via TDM, double-buffered --------
__global__ __launch_bounds__(256) void moe_down_kernel(const float* __restrict__ act,
                                                       const float* __restrict__ dw,
                                                       const int* __restrict__ cnt,
                                                       const int* __restrict__ off,
                                                       float* __restrict__ mbuf) {
  const int e = blockIdx.x, d0 = blockIdx.y * 128;
  const int count = cnt[e];
  if (count == 0) return;
  const int base = off[e];
  __shared__ __align__(16) float Af[2][64][36];
  __shared__ __align__(16) float Wf[2][128][36];
  const int tid = threadIdx.x, lane = tid & 31, wv = tid >> 5;
  const bool w0 = is_wave0();
  const int half = lane >> 4, nl = lane & 15;
  const size_t wbase = (size_t)e * DM * DFF;
  const float* wrow = &dw[wbase + (size_t)d0 * DFF];

  for (int mt = 0; mt * 64 < count; ++mt) {
    v8f acc[4];
#pragma unroll
    for (int i = 0; i < 4; ++i) acc[i] = {};

    auto stageA = [&](int buf, int k0) {
#pragma unroll
      for (int j = 0; j < 8; ++j) {
        int el = tid + 256 * j, r = el >> 5, c = el & 31;
        int ri = mt * 64 + r;
        Af[buf][r][c] = (ri < count) ? act[(size_t)(base + ri) * DFF + k0 + c] : 0.f;
      }
    };

    stageA(0, 0);
    if (w0) tdm_load_2d(wrow, &Wf[0][0][0], 128, DFF);
    const int KT = DFF >> 5;
    for (int kt = 0; kt < KT; ++kt) {
      const int cur = kt & 1;
      if (w0) wait_tensor();
      __syncthreads();
      if (kt + 1 < KT) {
        const int k0n = (kt + 1) << 5;
        stageA(cur ^ 1, k0n);
        if (w0) tdm_load_2d(wrow + k0n, &Wf[cur ^ 1][0][0], 128, DFF);
      }
      v16bf b = load_b_frag_f32(&Wf[cur][wv * 16][0], 36, lane);
#pragma unroll
      for (int i = 0; i < 4; ++i) {
        v16bf a = load_a_frag_f32(&Af[cur][i * 16][0], 36, lane);
        acc[i] = wmma_bf16(a, b, acc[i]);
      }
    }
    __syncthreads();
#pragma unroll
    for (int i = 0; i < 4; ++i)
#pragma unroll
      for (int r = 0; r < 8; ++r) {
        int ri = mt * 64 + i * 16 + r + 8 * half;
        if (ri < count)
          mbuf[(size_t)(base + ri) * DM + d0 + wv * 16 + nl] = acc[i][r];
      }
  }
}

// ---------------- Deterministic combine ----------------
__global__ __launch_bounds__(256) void combine_kernel(const float* __restrict__ x2,
                                                      const float* __restrict__ mbuf,
                                                      const int* __restrict__ tpair,
                                                      float* __restrict__ out) {
  const int t = blockIdx.x, tid = threadIdx.x;
  int p[TK];
#pragma unroll
  for (int s = 0; s < TK; ++s) p[s] = tpair[t * TK + s];
#pragma unroll
  for (int j = 0; j < 8; ++j) {
    int d = tid + 256 * j;
    float v = x2[(size_t)t * DM + d];
#pragma unroll
    for (int s = 0; s < TK; ++s) v += mbuf[(size_t)p[s] * DM + d];
    out[(size_t)t * DM + d] = v;
  }
}

extern "C" void kernel_launch(void* const* d_in, const int* in_sizes, int n_in,
                              void* d_out, int out_size, void* d_ws, size_t ws_size,
                              hipStream_t stream) {
  const float* x   = (const float*)d_in[0];
  const float* ln1 = (const float*)d_in[1];
  const float* qw  = (const float*)d_in[2];
  const float* kw  = (const float*)d_in[3];
  const float* vw  = (const float*)d_in[4];
  const float* qn  = (const float*)d_in[5];
  const float* kn  = (const float*)d_in[6];
  const float* ow  = (const float*)d_in[7];
  const float* ln2 = (const float*)d_in[8];
  const float* gw  = (const float*)d_in[9];
  const float* gup = (const float*)d_in[10];
  const float* dwn = (const float*)d_in[11];
  float* out = (float*)d_out;

  const size_t TD = (size_t)TT * DM;
  char* wsp = (char*)d_ws;
  auto alloc = [&](size_t bytes) {
    char* p = wsp;
    wsp += (bytes + 255) & ~(size_t)255;
    return p;
  };
  float* xn   = (float*)alloc(TD * 4);
  float* qb   = (float*)alloc(TD * 4);
  float* kb   = (float*)alloc(TD * 4);
  float* vb   = (float*)alloc(TD * 4);
  float* ctxb = (float*)alloc(TD * 4);
  float* x2   = (float*)alloc(TD * 4);
  float* x3   = (float*)alloc(TD * 4);
  float* actb = (float*)alloc((size_t)TT * TK * DFF * 4);
  float* mbuf = (float*)alloc((size_t)TT * TK * DM * 4);
  float* selw = (float*)alloc(TT * TK * 4);
  float* pwb  = (float*)alloc(TT * TK * 4);
  int* sele   = (int*)alloc(TT * TK * 4);
  int* cntb   = (int*)alloc(NE * 4);
  int* offb   = (int*)alloc(NE * 4);
  int* ptok   = (int*)alloc(TT * TK * 4);
  int* tpair  = (int*)alloc(TT * TK * 4);

  const dim3 gg(DM / 128, TT / 64);

  rmsnorm_kernel<<<TT, 256, 0, stream>>>(x, ln1, xn);
  gemm_nt_kernel<<<gg, 256, 0, stream>>>(xn, qw, nullptr, qb, TT, DM, DM);
  gemm_nt_kernel<<<gg, 256, 0, stream>>>(xn, kw, nullptr, kb, TT, DM, DM);
  gemm_nt_kernel<<<gg, 256, 0, stream>>>(xn, vw, nullptr, vb, TT, DM, DM);
  rmsnorm_kernel<<<TT, 256, 0, stream>>>(qb, qn, qb);
  rmsnorm_kernel<<<TT, 256, 0, stream>>>(kb, kn, kb);
  rope_kernel<<<TT, 1024, 0, stream>>>(qb, kb);
  attn_kernel<<<NH * 2, 256, 0, stream>>>(qb, kb, vb, ctxb);
  gemm_nt_kernel<<<gg, 256, 0, stream>>>(ctxb, ow, x, x2, TT, DM, DM);
  rmsnorm_kernel<<<TT, 256, 0, stream>>>(x2, ln2, x3);
  router_kernel<<<TT, 64, 0, stream>>>(x3, gw, sele, selw);
  build_lists_kernel<<<1, 64, 0, stream>>>(sele, selw, cntb, offb, ptok, pwb, tpair);
  moe_up_kernel<<<dim3(NE, DFF / 128), 256, 0, stream>>>(x3, gup, cntb, offb, ptok, pwb, actb);
  moe_down_kernel<<<dim3(NE, DM / 128), 256, 0, stream>>>(actb, dwn, cntb, offb, mbuf);
  combine_kernel<<<TT, 256, 0, stream>>>(x2, mbuf, tpair, out);
}
